// AttnBlockpp_9586367004817
// MI455X (gfx1250) — matrix operand order repaired and checked by reading, not compiled, not run on hardware
//
#include <hip/hip_runtime.h>
#include <hip/hip_bf16.h>

// ---------------------------------------------------------------------------
// AttnBlock for MI455X (gfx1250, wave32, WMMA bf16 16x16x32, f32 accum)
// B=8, C=256, H=W=64, N=4096, G=32 groups
// ---------------------------------------------------------------------------

#define BB 8
#define CC 256
#define NN 4096
#define GG 32
#define EPSV 1e-6f

typedef __bf16 bf16_t;
typedef bf16_t v16bf __attribute__((ext_vector_type(16)));
typedef bf16_t v8bf  __attribute__((ext_vector_type(8)));
typedef float  v8f   __attribute__((ext_vector_type(8)));

union BF16x16 { v16bf v; v8bf h[2]; };

static __device__ __forceinline__ v8bf ld8(const bf16_t* p) {
    return *(const v8bf*)p;
}

static __device__ __forceinline__ v8f wmma_bf16(v16bf a, v16bf b, v8f c) {
    // D = A(16x32 bf16) * B(32x16 bf16) + C(16x16 f32)
    return __builtin_amdgcn_wmma_f32_16x16x32_bf16(
        /*neg_a=*/false, a, /*neg_b=*/false, b,
        /*c_mod=*/(short)0, c, /*reuse_a=*/false, /*reuse_b=*/false);
}

// Async global -> LDS copy of 16 bytes (CDNA5 ASYNCcnt path, GV mode).
// LDS address = low 32 bits of the flat pointer (LDS aperture rule).
static __device__ __forceinline__ void async_cp16(void* lds_ptr, const void* gptr) {
    unsigned l = (unsigned)(unsigned long long)lds_ptr;
    asm volatile("global_load_async_to_lds_b128 %0, %1, off"
                 :: "v"(l), "v"(gptr) : "memory");
}

static __device__ __forceinline__ void wait_async0() {
    asm volatile("s_wait_asynccnt 0x0" ::: "memory");
}

// ---------------------------------------------------------------------------
// Kernel 0: convert the four C x C weight matrices to bf16.
// C^-0.5 = 1/16 is folded into W_q (so attention logits come out pre-scaled).
// Layout: wbf[4][C*C], order {q,k,v,o}.
// ---------------------------------------------------------------------------
__global__ __launch_bounds__(256) void wconv_kernel(
    const float* __restrict__ wq, const float* __restrict__ wk,
    const float* __restrict__ wv, const float* __restrict__ wo,
    bf16_t* __restrict__ wbf) {
    int i = blockIdx.x * 256 + threadIdx.x;          // 0 .. 4*65536-1
    const float* src[4] = {wq, wk, wv, wo};
    int m = i >> 16, r = i & 65535;
    float v = src[m][r];
    if (m == 0) v *= 0.0625f;                        // fold C^-0.5 into Wq
    wbf[i] = (bf16_t)v;
}

// ---------------------------------------------------------------------------
// Kernel 1: GroupNorm. One block per (b,g): 8 channels x 4096 pixels.
// Output hnT in [B][N][C] bf16 (c-contiguous) — the layout WMMA B/A
// fragments want for the projection GEMMs.
// ---------------------------------------------------------------------------
__global__ __launch_bounds__(256) void gn_kernel(
    const float* __restrict__ x, const float* __restrict__ gsc,
    const float* __restrict__ gbi, bf16_t* __restrict__ hnT) {
    __shared__ float rs[256], rss[256];
    int bg = blockIdx.x;
    int b = bg >> 5, g = bg & 31;
    const float* xg = x + ((size_t)b * CC + g * 8) * NN;  // 8 rows of N
    int ci = threadIdx.x & 7;         // channel-in-group (fixed per thread)
    int nb = threadIdx.x >> 3;        // pixel lane

    float s = 0.f, ss = 0.f;
    for (int n = nb; n < NN; n += 32) {
        float v = xg[(size_t)ci * NN + n];
        s += v; ss += v * v;
    }
    rs[threadIdx.x] = s; rss[threadIdx.x] = ss;
    __syncthreads();
    for (int st = 128; st > 0; st >>= 1) {
        if (threadIdx.x < st) {
            rs[threadIdx.x]  += rs[threadIdx.x + st];
            rss[threadIdx.x] += rss[threadIdx.x + st];
        }
        __syncthreads();
    }
    float mean = rs[0] * (1.0f / 32768.0f);
    float var  = rss[0] * (1.0f / 32768.0f) - mean * mean;
    float rinv = rsqrtf(var + EPSV);

    int c = g * 8 + ci;
    float a  = gsc[c] * rinv;
    float bb = gbi[c] - mean * a;
    bf16_t* dst = hnT + (size_t)b * NN * CC + c;
    for (int n = nb; n < NN; n += 32) {
        float v = xg[(size_t)ci * NN + n];
        dst[(size_t)n * CC] = (bf16_t)(v * a + bb);
    }
}

// ---------------------------------------------------------------------------
// Kernel 2: QKV projection GEMM.  out[oc,n] = sum_c W[oc,c] * hn[n,c] + bias
// grid: x = (N/128)*(C/128)=64 tiles, y = batch, z = mode {0:q,1:k,2:v}
// block: 256 threads = 8 waves laid out 4(oc) x 2(n); each wave owns a
// 32x64 output tile = 2x4 WMMA accumulators; K loop over C=256 in steps
// of 32 feeding v_wmma_f32_16x16x32_bf16 (fragments straight from L2).
// q,k stored transposed [B][N][C] bf16 (for attention fragments);
// v stored natural [B][C][N] bf16 (for the P*V B-operand).
// ---------------------------------------------------------------------------
__global__ __launch_bounds__(256) void qkv_kernel(
    const bf16_t* __restrict__ wbf, const float* __restrict__ bq,
    const float* __restrict__ bk, const float* __restrict__ bv,
    const bf16_t* __restrict__ hnT, bf16_t* __restrict__ qT,
    bf16_t* __restrict__ kT, bf16_t* __restrict__ vv) {
    int mode = blockIdx.z;
    int b = blockIdx.y;
    int nTile  = (blockIdx.x & 31) * 128;
    int ocTile = (blockIdx.x >> 5) * 128;
    int w = threadIdx.x >> 5, lane = threadIdx.x & 31;
    int half = lane >> 4, l16 = lane & 15;
    int ocW = ocTile + (w >> 1) * 32;
    int nW  = nTile  + (w & 1) * 64;

    const bf16_t* W    = wbf + (size_t)mode * (CC * CC);
    const bf16_t* Bsrc = hnT + (size_t)b * NN * CC;

    v8f acc[2][4] = {};
#pragma unroll
    for (int kk = 0; kk < CC; kk += 32) {
        BF16x16 a[2];
#pragma unroll
        for (int i = 0; i < 2; ++i) {
            const bf16_t* ap = W + (size_t)(ocW + i * 16 + l16) * CC + kk + half * 16;
            a[i].h[0] = ld8(ap);
            a[i].h[1] = ld8(ap + 8);
        }
        BF16x16 bfr[4];
#pragma unroll
        for (int j = 0; j < 4; ++j) {
            const bf16_t* bp = Bsrc + (size_t)(nW + j * 16 + l16) * CC + kk + half * 16;
            bfr[j].h[0] = ld8(bp);
            bfr[j].h[1] = ld8(bp + 8);
        }
#pragma unroll
        for (int i = 0; i < 2; ++i)
#pragma unroll
            for (int j = 0; j < 4; ++j)
                acc[i][j] = wmma_bf16(a[i].v, bfr[j].v, acc[i][j]);
    }

    if (mode < 2) {
        bf16_t* dst = (mode == 0) ? qT : kT;
        const float* bias = (mode == 0) ? bq : bk;
        float bscale = (mode == 0) ? 0.0625f : 1.0f;   // q bias also carries C^-0.5
#pragma unroll
        for (int i = 0; i < 2; ++i)
#pragma unroll
            for (int j = 0; j < 4; ++j) {
                int n   = nW + j * 16 + l16;
                int oc0 = ocW + i * 16 + 8 * half;
                v8bf pack;
#pragma unroll
                for (int r = 0; r < 8; ++r)
                    pack[r] = (bf16_t)(acc[i][j][r] + bias[oc0 + r] * bscale);
                *(v8bf*)(dst + ((size_t)b * NN + n) * CC + oc0) = pack;
            }
    } else {
#pragma unroll
        for (int i = 0; i < 2; ++i)
#pragma unroll
            for (int j = 0; j < 4; ++j) {
                int n = nW + j * 16 + l16;
#pragma unroll
                for (int r = 0; r < 8; ++r) {
                    int oc = ocW + i * 16 + 8 * half + r;
                    vv[((size_t)b * CC + oc) * NN + n] =
                        (bf16_t)(acc[i][j][r] + bv[oc]);
                }
            }
    }
}

// ---------------------------------------------------------------------------
// Kernel 3: flash attention.  grid (N/128, B); 8 waves, each owns 16 query
// rows and the full 16x256 f32 output accumulator (16 WMMA tiles).
// Q block staged once via async global->LDS (64 KB); K/V tiles of 32 keys
// double-buffered through the CDNA5 async-to-LDS pipe (ASYNCcnt), so the
// next slab streams in while the current one feeds the WMMAs.
// Online softmax with shuffle row-reductions; P tile goes D-layout -> LDS
// -> A-layout for the P*V WMMA (explicit s_wait_dscnt on the wave-local
// round trip).  Output stored bf16 [B][N][C] for the final projection.
// LDS: qs 64K + 2x(ks 16K + vs 16K) + ps 8K = 136 KB of the 320 KB WGP pool.
// ---------------------------------------------------------------------------
#define KS_ELEMS (32 * 256)
#define VS_ELEMS (256 * 32)

__global__ __launch_bounds__(256) void attn_kernel(
    const bf16_t* __restrict__ qT, const bf16_t* __restrict__ kT,
    const bf16_t* __restrict__ vv, bf16_t* __restrict__ Obf) {
    extern __shared__ char smem[];
    bf16_t* qs = (bf16_t*)smem;                  // [128][256]      64 KB
    bf16_t* ks = qs + 128 * 256;                 // [2][32][256]    32 KB
    bf16_t* vs = ks + 2 * KS_ELEMS;              // [2][256][32]    32 KB
    bf16_t* ps = vs + 2 * VS_ELEMS;              // [8][16][32]      8 KB

    int b = blockIdx.y;
    int qBase = blockIdx.x * 128;
    int tid = threadIdx.x, w = tid >> 5, lane = tid & 31;
    int half = lane >> 4, l16 = lane & 15;

    const bf16_t* Qg = qT + ((size_t)b * NN + qBase) * CC;
    const bf16_t* Kg = kT + (size_t)b * NN * CC;
    const bf16_t* Vg = vv + (size_t)b * CC * NN;

    // issue one K/V slab (32 keys) into LDS buffer bi via the async pipe
    auto stage_async = [&](int m0, int bi) {
        const bf16_t* ksrc = Kg + (size_t)m0 * CC;       // contiguous 16 KB
        bf16_t* kdst = ks + bi * KS_ELEMS;
        for (int u = tid; u < KS_ELEMS / 8; u += 256)
            async_cp16(kdst + u * 8, ksrc + u * 8);
        const bf16_t* vsrc = Vg + (size_t)tid * NN + m0; // row c=tid, 64 B
        bf16_t* vdst = vs + bi * VS_ELEMS + tid * 32;
#pragma unroll
        for (int u = 0; u < 4; ++u)
            async_cp16(vdst + u * 8, vsrc + u * 8);
    };

    // prologue: stage Q block + first K/V slab asynchronously
    for (int u = tid; u < 128 * CC / 8; u += 256)
        async_cp16(qs + u * 8, Qg + u * 8);
    stage_async(0, 0);

    v8f Oacc[16] = {};
    float mrun[8], lrun[8];
#pragma unroll
    for (int r = 0; r < 8; ++r) { mrun[r] = -3.0e38f; lrun[r] = 0.f; }

    wait_async0();
    __syncthreads();

    for (int m0 = 0; m0 < NN; m0 += 32) {
        int bi = (m0 >> 5) & 1;
        bool more = (m0 + 32) < NN;
        // prefetch the next slab into the other buffer while we compute;
        // safe: every wave passed the previous barrier, so buffer bi^1 is idle
        if (more) stage_async(m0 + 32, bi ^ 1);

        const bf16_t* ksb = ks + bi * KS_ELEMS;
        const bf16_t* vsb = vs + bi * VS_ELEMS;

        // S = Q^T K for two 16-key columns (logits pre-scaled via Wq)
        v8f s0 = {}, s1 = {};
#pragma unroll
        for (int kk = 0; kk < CC; kk += 32) {
            BF16x16 aq;
            const bf16_t* ap = qs + (size_t)(w * 16 + l16) * CC + kk + half * 16;
            aq.h[0] = ld8(ap); aq.h[1] = ld8(ap + 8);
            BF16x16 b0, b1;
            const bf16_t* bp0 = ksb + (size_t)l16 * CC + kk + half * 16;
            const bf16_t* bp1 = ksb + (size_t)(16 + l16) * CC + kk + half * 16;
            b0.h[0] = ld8(bp0); b0.h[1] = ld8(bp0 + 8);
            b1.h[0] = ld8(bp1); b1.h[1] = ld8(bp1 + 8);
            s0 = wmma_bf16(aq.v, b0.v, s0);
            s1 = wmma_bf16(aq.v, b1.v, s1);
        }

        // online softmax per query row (row = r + 8*half, cols across lanes)
#pragma unroll
        for (int r = 0; r < 8; ++r) {
            float mx = fmaxf(s0[r], s1[r]);
#pragma unroll
            for (int off = 1; off < 16; off <<= 1)
                mx = fmaxf(mx, __shfl_xor(mx, off, 32));
            float nm = fmaxf(mrun[r], mx);
            float corr = __expf(mrun[r] - nm);
            mrun[r] = nm;
            float p0 = __expf(s0[r] - nm);
            float p1 = __expf(s1[r] - nm);
            float rsum = p0 + p1;
#pragma unroll
            for (int off = 1; off < 16; off <<= 1)
                rsum += __shfl_xor(rsum, off, 32);
            lrun[r] = lrun[r] * corr + rsum;
#pragma unroll
            for (int ct = 0; ct < 16; ++ct) Oacc[ct][r] *= corr;
            bf16_t* pw = ps + (size_t)(w * 16 + r + 8 * half) * 32;
            pw[l16]      = (bf16_t)p0;
            pw[16 + l16] = (bf16_t)p1;
        }
        asm volatile("s_wait_dscnt 0" ::: "memory");  // wave-local LDS RAW

        // O += P * V^T   (A = P 16x32 from LDS, B = V slab columns)
        BF16x16 pf;
        const bf16_t* pp = ps + (size_t)(w * 16 + l16) * 32 + half * 16;
        pf.h[0] = ld8(pp); pf.h[1] = ld8(pp + 8);
#pragma unroll
        for (int ct = 0; ct < 16; ++ct) {
            BF16x16 vf;
            const bf16_t* vp = vsb + (size_t)(ct * 16 + l16) * 32 + half * 16;
            vf.h[0] = ld8(vp); vf.h[1] = ld8(vp + 8);
            Oacc[ct] = wmma_bf16(pf.v, vf.v, Oacc[ct]);
        }

        // fence the prefetch and rejoin before swapping buffers
        if (more) wait_async0();
        __syncthreads();
    }

    // normalize and store bf16 [B][N][C]
#pragma unroll
    for (int r = 0; r < 8; ++r) {
        float inv = 1.0f / lrun[r];
        size_t row = (size_t)b * NN + qBase + w * 16 + r + 8 * half;
#pragma unroll
        for (int ct = 0; ct < 16; ++ct)
            Obf[row * CC + ct * 16 + l16] = (bf16_t)(Oacc[ct][r] * inv);
    }
}

// ---------------------------------------------------------------------------
// Kernel 4: output projection + residual:  out = (x + Wo*attn + bo)/sqrt(2)
// Same WMMA tiling as the QKV GEMM; f32 output in the reference's
// [B][C][H][W] layout.
// ---------------------------------------------------------------------------
__global__ __launch_bounds__(256) void proj_kernel(
    const bf16_t* __restrict__ wo_bf, const float* __restrict__ bo,
    const bf16_t* __restrict__ Obf, const float* __restrict__ x,
    float* __restrict__ out) {
    int b = blockIdx.y;
    int nTile  = (blockIdx.x & 31) * 128;
    int ocTile = (blockIdx.x >> 5) * 128;
    int w = threadIdx.x >> 5, lane = threadIdx.x & 31;
    int half = lane >> 4, l16 = lane & 15;
    int ocW = ocTile + (w >> 1) * 32;
    int nW  = nTile  + (w & 1) * 64;

    const bf16_t* Bsrc = Obf + (size_t)b * NN * CC;

    v8f acc[2][4] = {};
#pragma unroll
    for (int kk = 0; kk < CC; kk += 32) {
        BF16x16 a[2];
#pragma unroll
        for (int i = 0; i < 2; ++i) {
            const bf16_t* ap = wo_bf + (size_t)(ocW + i * 16 + l16) * CC + kk + half * 16;
            a[i].h[0] = ld8(ap);
            a[i].h[1] = ld8(ap + 8);
        }
        BF16x16 bfr[4];
#pragma unroll
        for (int j = 0; j < 4; ++j) {
            const bf16_t* bp = Bsrc + (size_t)(nW + j * 16 + l16) * CC + kk + half * 16;
            bfr[j].h[0] = ld8(bp);
            bfr[j].h[1] = ld8(bp + 8);
        }
#pragma unroll
        for (int i = 0; i < 2; ++i)
#pragma unroll
            for (int j = 0; j < 4; ++j)
                acc[i][j] = wmma_bf16(a[i].v, bfr[j].v, acc[i][j]);
    }

    const float inv_sqrt2 = 0.70710678118654752f;
#pragma unroll
    for (int i = 0; i < 2; ++i)
#pragma unroll
        for (int j = 0; j < 4; ++j) {
            int n = nW + j * 16 + l16;
#pragma unroll
            for (int r = 0; r < 8; ++r) {
                int oc = ocW + i * 16 + 8 * half + r;
                size_t idx = ((size_t)b * CC + oc) * NN + n;
                out[idx] = (x[idx] + acc[i][j][r] + bo[oc]) * inv_sqrt2;
            }
        }
}

// ---------------------------------------------------------------------------
// Host-side launch
// ---------------------------------------------------------------------------
extern "C" void kernel_launch(void* const* d_in, const int* in_sizes, int n_in,
                              void* d_out, int out_size, void* d_ws, size_t ws_size,
                              hipStream_t stream) {
    (void)in_sizes; (void)n_in; (void)out_size; (void)ws_size;
    const float* x        = (const float*)d_in[0];
    const float* gn_scale = (const float*)d_in[1];
    const float* gn_bias  = (const float*)d_in[2];
    const float* w_q      = (const float*)d_in[3];
    const float* b_q      = (const float*)d_in[4];
    const float* w_k      = (const float*)d_in[5];
    const float* b_k      = (const float*)d_in[6];
    const float* w_v      = (const float*)d_in[7];
    const float* b_v      = (const float*)d_in[8];
    const float* w_o      = (const float*)d_in[9];
    const float* b_o      = (const float*)d_in[10];
    float* out = (float*)d_out;

    const size_t SZ = (size_t)BB * NN * CC * sizeof(bf16_t);  // 16 MB per tensor
    char* ws = (char*)d_ws;
    bf16_t* hnT = (bf16_t*)(ws);
    bf16_t* qT  = (bf16_t*)(ws + 1 * SZ);
    bf16_t* kT  = (bf16_t*)(ws + 2 * SZ);
    bf16_t* vv  = (bf16_t*)(ws + 3 * SZ);
    bf16_t* Obf = (bf16_t*)(ws + 4 * SZ);
    bf16_t* wbf = (bf16_t*)(ws + 5 * SZ);   // 4 x C*C bf16

    wconv_kernel<<<dim3(4 * CC * CC / 256), 256, 0, stream>>>(w_q, w_k, w_v, w_o, wbf);
    gn_kernel<<<dim3(BB * GG), 256, 0, stream>>>(x, gn_scale, gn_bias, hnT);
    qkv_kernel<<<dim3(64, BB, 3), 256, 0, stream>>>(wbf, b_q, b_k, b_v, hnT, qT, kT, vv);

    size_t smem = (size_t)(128 * 256 + 2 * KS_ELEMS + 2 * VS_ELEMS + 8 * 16 * 32)
                  * sizeof(bf16_t);   // 136 KB
    attn_kernel<<<dim3(NN / 128, BB), 256, smem, stream>>>(qT, kT, vv, Obf);

    proj_kernel<<<dim3(64, BB), 256, 0, stream>>>(wbf + 3 * CC * CC, b_o, Obf, x, out);
}
